// FunctionalAttention_11166914970236
// MI455X (gfx1250) — matrix operand order repaired and checked by reading, not compiled, not run on hardware
//
#include <hip/hip_runtime.h>
#include <type_traits>

typedef __attribute__((ext_vector_type(16))) _Float16 v16h;
typedef __attribute__((ext_vector_type(8)))  _Float16 v8h;
typedef __attribute__((ext_vector_type(2)))  _Float16 v2h;
typedef __attribute__((ext_vector_type(8)))  float    v8f;

#define T_SEQ 2048
#define DHEAD 64
#define NHEAD 16

// LDS pitches in halfwords: rows 16B-aligned, conflict-free stride-4 bank tilings
#define KP 72   // 144 B rows
#define VP 40   //  80 B rows
#define PP 24   //  48 B rows

// One-instruction max: avoids compiler-inserted canonicalize (v_max v,v,v) fixups.
__device__ __forceinline__ float vmaxf(float a, float b) {
    float d;
    asm("v_max_num_f32_e32 %0, %1, %2" : "=v"(d) : "v"(a), "v"(b));
    return d;
}

// Fused DPP16 ROW_XMASK reduction steps: one VOP2_DPP each, VALU-only.
#define DPP_MAX_STEP(y, m)                                                      \
    { float _t;                                                                 \
      asm("v_max_num_f32_dpp %0, %1, %1 row_xmask:" #m                          \
          " row_mask:0xf bank_mask:0xf bound_ctrl:1"                            \
          : "=v"(_t) : "v"(y));                                                 \
      y = _t; }
#define DPP_ADD_STEP(y, m)                                                      \
    { float _t;                                                                 \
      asm("v_add_f32_dpp %0, %1, %1 row_xmask:" #m                              \
          " row_mask:0xf bank_mask:0xf bound_ctrl:1"                            \
          : "=v"(_t) : "v"(y));                                                 \
      y = _t; }

__device__ __forceinline__ float row_max16(float x) {
    DPP_MAX_STEP(x, 1) DPP_MAX_STEP(x, 2) DPP_MAX_STEP(x, 4) DPP_MAX_STEP(x, 8)
    return x;
}
__device__ __forceinline__ float row_sum16(float x) {
    DPP_ADD_STEP(x, 1) DPP_ADD_STEP(x, 2) DPP_ADD_STEP(x, 4) DPP_ADD_STEP(x, 8)
    return x;
}

__device__ __forceinline__ float fast_exp2(float x) {
#if __has_builtin(__builtin_amdgcn_exp2f)
    return __builtin_amdgcn_exp2f(x);   // raw v_exp_f32
#else
    return exp2f(x);
#endif
}

// v_cvt_pk_rtz_f16_f32: two f32 -> packed 2xf16 in one instruction
__device__ __forceinline__ v2h pk_h2(float a, float b) {
    return __builtin_bit_cast(v2h, __builtin_amdgcn_cvt_pkrtz(a, b));
}

__global__ __launch_bounds__(128)
void fa_fwd_wmma(const float* __restrict__ qg, const float* __restrict__ kg,
                 const float* __restrict__ vg, float* __restrict__ out)
{
    __shared__ __align__(16) _Float16 kbuf[2][32][KP];      // K block, [key][d]
    __shared__ __align__(16) _Float16 vbufT[2][DHEAD][VP];  // V block, [d][key]
    __shared__ __align__(16) _Float16 pbuf[4][16][PP];      // per-wave P relayout

    const int tid  = threadIdx.x;
    const int wave = tid >> 5;
    const int lane = tid & 31;
    const int half = lane >> 4;
    const int colN = lane & 15;

    const int bh = blockIdx.y;
    const int b  = bh >> 4;
    const int h  = bh & 15;
    const int qgroup = blockIdx.x;
    const int q0w    = qgroup * 64 + wave * 16;
    const int qhi    = q0w + 15;

    const size_t baseBH = (size_t)b * T_SEQ * DHEAD * NHEAD + (size_t)h;

    // ---- Q fragments (A layout), 1/d and log2(e) folded in ----
    v16h aq[2];
    {
        const float qsc = 1.4426950408889634f / 64.0f;   // log2(e)/d
        const int qrow = q0w + colN;
        const size_t qbase = baseBH + (size_t)qrow * DHEAD * NHEAD;
        #pragma unroll
        for (int c = 0; c < 2; ++c) {
            union { v16h v; v2h p[8]; } ua;
            #pragma unroll
            for (int r = 0; r < 8; ++r) {
                const int k0 = half * 16 + 2 * r;
                const float e0 = qg[qbase + (size_t)(32 * c + k0) * NHEAD] * qsc;
                const float e1 = qg[qbase + (size_t)(32 * c + k0 + 1) * NHEAD] * qsc;
                ua.p[r] = pk_h2(e0, e1);
            }
            aq[c] = ua.v;
        }
    }

    float m_i[8], l_i[8];
    v8f acc[4];
    const v8f vzero = {};
    #pragma unroll
    for (int r = 0; r < 8; ++r) { m_i[r] = -1e30f; l_i[r] = 0.0f; }
    #pragma unroll
    for (int n = 0; n < 4; ++n) acc[n] = vzero;

    const int nblk = 2 * (qgroup + 1);
    float kf[16], vf[16];

    // ---- prologue: stage block 0 ----
    #pragma unroll
    for (int i = 0; i < 8; ++i) {
        const int p = i * 128 + tid;
        const int key = p >> 5, d2 = (p & 31) * 2;
        const size_t g = baseBH + (size_t)key * 1024 + (size_t)d2 * 16;
        kf[2 * i] = kg[g]; kf[2 * i + 1] = kg[g + 16];
        const int dv = p >> 4, k2 = (p & 15) * 2;
        const size_t gv = baseBH + (size_t)k2 * 1024 + (size_t)dv * 16;
        vf[2 * i] = vg[gv]; vf[2 * i + 1] = vg[gv + 1024];
    }
    #pragma unroll
    for (int i = 0; i < 8; ++i) {
        const int p = i * 128 + tid;
        const int key = p >> 5, d2 = (p & 31) * 2;
        *(v2h*)&kbuf[0][key][d2] = pk_h2(kf[2 * i], kf[2 * i + 1]);
        const int dv = p >> 4, k2 = (p & 15) * 2;
        *(v2h*)&vbufT[0][dv][k2] = pk_h2(vf[2 * i], vf[2 * i + 1]);
    }
    __syncthreads();

    int cur = 0;
    for (int j = 0; j < nblk; ++j) {
        const int kbase = j * 32;
        const bool more = (j + 1 < nblk);

        // ---- issue next block's global loads (overlap with WMMA below) ----
        if (more) {
            const int nb = kbase + 32;
            #pragma unroll
            for (int i = 0; i < 8; ++i) {
                const int p = i * 128 + tid;
                const int key = p >> 5, d2 = (p & 31) * 2;
                const size_t g = baseBH + (size_t)(nb + key) * 1024 + (size_t)d2 * 16;
                kf[2 * i] = kg[g]; kf[2 * i + 1] = kg[g + 16];
                const int dv = p >> 4, k2 = (p & 15) * 2;
                const size_t gv = baseBH + (size_t)(nb + k2) * 1024 + (size_t)dv * 16;
                vf[2 * i] = vg[gv]; vf[2 * i + 1] = vg[gv + 1024];
            }
        }

        // wave-uniform: branch scalar, keep EXEC all-ones around WMMA
        if (__builtin_amdgcn_readfirstlane(kbase <= qhi ? 1 : 0)) {
            // ---- S = Q*K^T : 4 WMMAs, fragments as 2x ds_load_b128 each ----
            v8f s[2];
            s[0] = vzero; s[1] = vzero;
            #pragma unroll
            for (int n = 0; n < 2; ++n)
                #pragma unroll
                for (int c = 0; c < 2; ++c) {
                    union { v16h v; v8h hh[2]; } ub;
                    const v8h* kr = (const v8h*)&kbuf[cur][n * 16 + colN][32 * c + 16 * half];
                    ub.hh[0] = kr[0];
                    ub.hh[1] = kr[1];
                    s[n] = __builtin_amdgcn_wmma_f32_16x16x32_f16(
                        false, aq[c], false, ub.v, (short)0, s[n], false, false);
                }

            // ---- online softmax (log2 domain, fused-DPP row reductions) ----
            const auto softmax_upd = [&](auto maskc) {
                constexpr bool MASK = decltype(maskc)::value;
                #pragma unroll
                for (int r = 0; r < 8; ++r) {
                    const int row  = 8 * half + r;
                    float f0 = s[0][r];
                    float f1 = s[1][r];
                    if (MASK) {
                        const int qrow = q0w + row;
                        if (kbase + colN      > qrow) f0 = -1e30f;
                        if (kbase + 16 + colN > qrow) f1 = -1e30f;
                    }
                    const float rmax  = row_max16(vmaxf(f0, f1));
                    const float mnew  = vmaxf(m_i[r], rmax);
                    const float alpha = fast_exp2(m_i[r] - mnew);
                    const float p0 = fast_exp2(f0 - mnew);
                    const float p1 = fast_exp2(f1 - mnew);
                    const float rsum = row_sum16(p0 + p1);
                    l_i[r] = l_i[r] * alpha + rsum;
                    m_i[r] = mnew;
                    #pragma unroll
                    for (int n = 0; n < 4; ++n) acc[n][r] *= alpha;
                    const v2h ph = pk_h2(p0, p1);
                    pbuf[wave][row][colN]      = ph[0];
                    pbuf[wave][row][16 + colN] = ph[1];
                }
            };
            if (__builtin_amdgcn_readfirstlane((kbase + 31) > q0w ? 1 : 0))
                softmax_upd(std::integral_constant<bool, true>{});
            else
                softmax_upd(std::integral_constant<bool, false>{});

            asm volatile("" ::: "memory");
#if __has_builtin(__builtin_amdgcn_wave_barrier)
            __builtin_amdgcn_wave_barrier();
#endif

            // ---- O += P*V : 4 WMMAs ----
            union { v16h v; v8h hh[2]; } up;
            up.hh[0] = *(const v8h*)&pbuf[wave][colN][16 * half];
            up.hh[1] = *(const v8h*)&pbuf[wave][colN][16 * half + 8];
            #pragma unroll
            for (int n = 0; n < 4; ++n) {
                union { v16h v; v8h hh[2]; } uv;
                const v8h* vr = (const v8h*)&vbufT[cur][n * 16 + colN][16 * half];
                uv.hh[0] = vr[0];
                uv.hh[1] = vr[1];
                acc[n] = __builtin_amdgcn_wmma_f32_16x16x32_f16(
                    false, up.v, false, uv.v, (short)0, acc[n], false, false);
            }
        }

        // ---- convert + store next block into the other buffer ----
        if (more) {
            const int nxt = cur ^ 1;
            #pragma unroll
            for (int i = 0; i < 8; ++i) {
                const int p = i * 128 + tid;
                const int key = p >> 5, d2 = (p & 31) * 2;
                *(v2h*)&kbuf[nxt][key][d2] = pk_h2(kf[2 * i], kf[2 * i + 1]);
                const int dv = p >> 4, k2 = (p & 15) * 2;
                *(v2h*)&vbufT[nxt][dv][k2] = pk_h2(vf[2 * i], vf[2 * i + 1]);
            }
        }
        __syncthreads();
        cur ^= 1;
    }

    // ---- epilogue: normalize, scatter to [b, t, d, h] ----
    #pragma unroll
    for (int r = 0; r < 8; ++r) {
        const int qrow = q0w + 8 * half + r;
        const float inv = 1.0f / l_i[r];
        const size_t obase = baseBH + (size_t)qrow * DHEAD * NHEAD;
        #pragma unroll
        for (int n = 0; n < 4; ++n)
            out[obase + (size_t)(n * 16 + colN) * NHEAD] = acc[n][r] * inv;
    }
}

extern "C" void kernel_launch(void* const* d_in, const int* in_sizes, int n_in,
                              void* d_out, int out_size, void* d_ws, size_t ws_size,
                              hipStream_t stream) {
    (void)in_sizes; (void)n_in; (void)d_ws; (void)ws_size; (void)out_size;
    const float* q = (const float*)d_in[0];
    const float* k = (const float*)d_in[1];
    const float* v = (const float*)d_in[2];
    float* o = (float*)d_out;

    dim3 grid(T_SEQ / 64, 4 * NHEAD);   // (query groups of 64, B*H)
    dim3 block(128);                    // 4 wave32 waves
    fa_fwd_wmma<<<grid, block, 0, stream>>>(q, k, v, o);
}
